// Long_encoder_42649025249788
// MI455X (gfx1250) — compile-verified, matrix operand order checked by reading, not run on hardware
//
#include <hip/hip_runtime.h>
#include <hip/hip_bf16.h>
#include <math.h>

// ---------------- CDNA5 WMMA types ----------------
typedef __attribute__((ext_vector_type(16))) _Float16 v16h;
typedef __attribute__((ext_vector_type(8)))  _Float16 h8;
typedef __attribute__((ext_vector_type(8)))  float    v8f;

// Problem constants (from reference)
#define NSEQ   224      // B*C = 32*7
#define NTOK   1023     // (8192-16)/8+1
#define DM     128
#define DI     256
#define DS     16
#define DTR    8
#define DFF    256
#define LDXD   48       // padded leading dim for x_dbl (40 -> 48)

// f16 weight arena (halves), per layer:
//  in_proj 128x512      @      0 (65536)
//  out_proj 256x128     @  65536 (32768)
//  lin1 128x256         @  98304 (32768)
//  lin2 256x128         @ 131072 (32768)
//  x_proj 256x40 pad48  @ 163840 (12288)
#define WH_LAYER  176128
#define WH_TOTAL  (3 * WH_LAYER)     // 528384 halves = 264192 floats

// K-index map for 16-bit A/B fragments of v_wmma_f32_16x16x32_f16 (wave32).
// lanes 0-15 hold K=0..7 (VGPR0..3, pairs) and K=16..23 (VGPR4..7);
// lanes 16-31 hold K=8..15 and K=24..31.  B assumed symmetric (lane = column).
__device__ __forceinline__ int kmap(int i, int j, int half) {
    return ((i >= 4) ? 16 : 0) + 8 * half + 2 * (i & 3) + j;
}

// ---------------------------------------------------------------------------
// FAST GEMM: C[MxNv] = act(A[MxKT] @ W + bias), W pre-converted to f16
// col-major (Wh[col*KT + k]).  Requires M%16==0, Nv%(16*NT)==0, lda%4==0.
// One wave = 16x(16*NT) strip; K fully unrolled; A loads 4xb128,
// each B fragment 2xb128.  MODE: 0=none, 1=+bias, 2=+bias+exact GELU.
// ---------------------------------------------------------------------------
template<int MODE, int KT, int NT>
__global__ __launch_bounds__(256)
void gemm_wmma_fast(const float* __restrict__ A, const _Float16* __restrict__ Wh,
                    const float* __restrict__ bias, float* __restrict__ C,
                    int M, int Nv, int lda, int ldc) {
    const int wave = threadIdx.x >> 5;
    const int lane = threadIdx.x & 31;
    const int r    = lane & 15;
    const int half = lane >> 4;

    const int ntm  = M >> 4;
    const int ntn  = Nv / (16 * NT);
    const int tile = blockIdx.x * 8 + wave;
    if (tile >= ntm * ntn) return;             // wave-uniform
    const int tm  = tile / ntn;
    const int tn0 = (tile % ntn) * NT;
    const int row = tm * 16 + r;

    const float*    Arow = A  + (size_t)row * lda + 8 * half;
    const _Float16* Wc   = Wh + (size_t)(tn0 * 16 + r) * KT + 8 * half;

    v8f zero = {};
    v8f acc[NT];
#pragma unroll
    for (int t = 0; t < NT; ++t) acc[t] = zero;

#pragma unroll
    for (int k0 = 0; k0 < KT; k0 += 32) {
        const float4* A4 = (const float4*)(Arow + k0);
        float4 a0 = A4[0], a1 = A4[1];         // k = k0 + 8*half + 0..7
        float4 a2 = A4[4], a3 = A4[5];         // k = k0 + 8*half + 16..23
        v16h a;
        a[0] = (_Float16)a0.x; a[1] = (_Float16)a0.y; a[2] = (_Float16)a0.z; a[3] = (_Float16)a0.w;
        a[4] = (_Float16)a1.x; a[5] = (_Float16)a1.y; a[6] = (_Float16)a1.z; a[7] = (_Float16)a1.w;
        a[8]  = (_Float16)a2.x; a[9]  = (_Float16)a2.y; a[10] = (_Float16)a2.z; a[11] = (_Float16)a2.w;
        a[12] = (_Float16)a3.x; a[13] = (_Float16)a3.y; a[14] = (_Float16)a3.z; a[15] = (_Float16)a3.w;
#pragma unroll
        for (int t = 0; t < NT; ++t) {
            const h8* Bp = (const h8*)(Wc + (size_t)t * 16 * KT + k0);
            h8 b0 = Bp[0];                     // k = k0 + 8*half + 0..7
            h8 b1 = Bp[2];                     // k = k0 + 8*half + 16..23
            v16h b;
#pragma unroll
            for (int m = 0; m < 8; ++m) { b[m] = b0[m]; b[8 + m] = b1[m]; }
            acc[t] = __builtin_amdgcn_wmma_f32_16x16x32_f16(
                false, a, false, b, (short)0, acc[t], false, false);
        }
    }

#pragma unroll
    for (int t = 0; t < NT; ++t) {
        int col = (tn0 + t) * 16 + r;
        float bv = (MODE >= 1) ? bias[col] : 0.f;
#pragma unroll
        for (int v = 0; v < 8; ++v) {
            int orow = tm * 16 + v + 8 * half;
            float y = acc[t][v] + bv;
            if (MODE == 2) y = 0.5f * y * (1.f + erff(y * 0.70710678118f));
            C[(size_t)orow * ldc + col] = y;
        }
    }
}

// ---------------------------------------------------------------------------
// Guarded generic GEMM (f32 weights, any shape) -- small-ws fallback only
// ---------------------------------------------------------------------------
template<int MODE>
__global__ __launch_bounds__(256)
void gemm_wmma(const float* __restrict__ A, const float* __restrict__ W,
               const float* __restrict__ bias, float* __restrict__ C,
               int M, int Nv, int K, int lda, int ldw, int ldc) {
    const int wave = threadIdx.x >> 5;
    const int lane = threadIdx.x & 31;
    const int r    = lane & 15;
    const int half = lane >> 4;

    const int ntm  = (M + 15) >> 4;
    const int ntn4 = (Nv + 63) >> 6;
    const int tile = blockIdx.x * 8 + wave;
    if (tile >= ntm * ntn4) return;
    const int tm  = tile / ntn4;
    const int tn0 = (tile % ntn4) * 4;
    const int row = tm * 16 + r;

    v8f acc0 = {}, acc1 = {}, acc2 = {}, acc3 = {};

    for (int k0 = 0; k0 < K; k0 += 32) {
        v16h a;
#pragma unroll
        for (int i = 0; i < 8; ++i)
#pragma unroll
            for (int j = 0; j < 2; ++j) {
                int k = k0 + kmap(i, j, half);
                float v = (k < K && row < M) ? A[(size_t)row * lda + k] : 0.f;
                a[2 * i + j] = (_Float16)v;
            }
#pragma unroll
        for (int t = 0; t < 4; ++t) {
            int col = (tn0 + t) * 16 + r;
            v16h b;
#pragma unroll
            for (int i = 0; i < 8; ++i)
#pragma unroll
                for (int j = 0; j < 2; ++j) {
                    int k = k0 + kmap(i, j, half);
                    float v = (k < K && col < Nv) ? W[(size_t)k * ldw + col] : 0.f;
                    b[2 * i + j] = (_Float16)v;
                }
            v8f* accp = (t == 0) ? &acc0 : (t == 1) ? &acc1 : (t == 2) ? &acc2 : &acc3;
            *accp = __builtin_amdgcn_wmma_f32_16x16x32_f16(
                false, a, false, b, (short)0, *accp, false, false);
        }
    }

#pragma unroll
    for (int t = 0; t < 4; ++t) {
        int col = (tn0 + t) * 16 + r;
        if (col >= Nv) continue;
        v8f accv = (t == 0) ? acc0 : (t == 1) ? acc1 : (t == 2) ? acc2 : acc3;
        float bv = (MODE >= 1) ? bias[col] : 0.f;
#pragma unroll
        for (int v = 0; v < 8; ++v) {
            int orow = tm * 16 + v + 8 * half;
            if (orow >= M) continue;
            float y = accv[v] + bv;
            if (MODE == 2) y = 0.5f * y * (1.f + erff(y * 0.70710678118f));
            C[(size_t)orow * ldc + col] = y;
        }
    }
}

// ---------------------------------------------------------------------------
// Patch extraction + projection, fused gather-GEMM (K=16 zero-padded to 32)
// ---------------------------------------------------------------------------
__global__ __launch_bounds__(256)
void proj_wmma(const float* __restrict__ z, const float* __restrict__ Wp,
               const float* __restrict__ bp, float* __restrict__ X,
               int s0, int Mc) {
    const int wave = threadIdx.x >> 5;
    const int lane = threadIdx.x & 31;
    const int r    = lane & 15;
    const int half = lane >> 4;

    const int ntm  = (Mc + 15) >> 4;
    const int tile = blockIdx.x * 8 + wave;
    if (tile >= ntm * 2) return;
    const int tm  = tile >> 1;
    const int tn0 = (tile & 1) * 4;
    const int row = tm * 16 + r;

    v8f acc0 = {}, acc1 = {}, acc2 = {}, acc3 = {};

    v16h a;
#pragma unroll
    for (int i = 0; i < 8; ++i)
#pragma unroll
        for (int j = 0; j < 2; ++j) {
            int k = kmap(i, j, half);
            float v = 0.f;
            if (k < 16 && row < Mc) {
                int sl = row / NTOK;
                int n  = row - sl * NTOK;
                v = z[(size_t)(s0 + sl) * 8192 + n * 8 + k];
            }
            a[2 * i + j] = (_Float16)v;
        }
#pragma unroll
    for (int t = 0; t < 4; ++t) {
        int col = (tn0 + t) * 16 + r;
        v16h b;
#pragma unroll
        for (int i = 0; i < 8; ++i)
#pragma unroll
            for (int j = 0; j < 2; ++j) {
                int k = kmap(i, j, half);
                float v = (k < 16) ? Wp[k * DM + col] : 0.f;
                b[2 * i + j] = (_Float16)v;
            }
        v8f* accp = (t == 0) ? &acc0 : (t == 1) ? &acc1 : (t == 2) ? &acc2 : &acc3;
        *accp = __builtin_amdgcn_wmma_f32_16x16x32_f16(
            false, a, false, b, (short)0, *accp, false, false);
    }

#pragma unroll
    for (int t = 0; t < 4; ++t) {
        int col = (tn0 + t) * 16 + r;
        v8f accv = (t == 0) ? acc0 : (t == 1) ? acc1 : (t == 2) ? acc2 : acc3;
        float bv = bp[col];
#pragma unroll
        for (int v = 0; v < 8; ++v) {
            int orow = tm * 16 + v + 8 * half;
            if (orow >= Mc) continue;
            X[(size_t)orow * DM + col] = accv[v] + bv;
        }
    }
}

// ---------------------------------------------------------------------------
// Weight f32 row-major [K x Nsrc] -> f16 col-major [Npad x K], zero-padded
// ---------------------------------------------------------------------------
__global__ void convert_w(const float* __restrict__ W, _Float16* __restrict__ Wh,
                          int K, int Nsrc, int Npad) {
    int idx = blockIdx.x * blockDim.x + threadIdx.x;
    if (idx >= K * Npad) return;
    int col = idx / K, k = idx - col * K;
    Wh[idx] = (col < Nsrc) ? (_Float16)W[(size_t)k * Nsrc + col] : (_Float16)0.f;
}

// ---------------------------------------------------------------------------
// Causal depthwise conv (D_CONV=4) + SiLU on xc = xz[:, 0:256]
// ---------------------------------------------------------------------------
__global__ void conv_silu(const float* __restrict__ xz, const float* __restrict__ cw,
                          const float* __restrict__ cb, float* __restrict__ xc, int Mc) {
    int idx = blockIdx.x * blockDim.x + threadIdx.x;
    if (idx >= Mc * DI) return;
    int t = idx >> 8, d = idx & (DI - 1);
    int n = t % NTOK;
    float acc = cb[d];
#pragma unroll
    for (int k = 0; k < 4; ++k) {
        int off = 3 - k;
        if (n - off >= 0) acc += xz[(size_t)(t - off) * (2 * DI) + d] * cw[d * 4 + k];
    }
    xc[idx] = acc / (1.f + expf(-acc));          // silu
}

// ---------------------------------------------------------------------------
// dt = softplus(x_dbl[:, 0:8] @ dt_proj_w + dt_proj_b)
// ---------------------------------------------------------------------------
__global__ void dt_softplus(const float* __restrict__ xdbl, const float* __restrict__ dtw,
                            const float* __restrict__ dtb, float* __restrict__ dt, int Mc) {
    int idx = blockIdx.x * blockDim.x + threadIdx.x;
    if (idx >= Mc * DI) return;
    int t = idx >> 8, d = idx & (DI - 1);
    float acc = dtb[d];
#pragma unroll
    for (int r = 0; r < DTR; ++r)
        acc += xdbl[(size_t)t * LDXD + r] * dtw[r * DI + d];
    dt[idx] = (acc > 20.f) ? acc : log1pf(expf(acc));
}

// ---------------------------------------------------------------------------
// Selective scan with fused SiLU gating.  One block = one sequence; thread d
// keeps its 16-state in VGPRs; B/C staged 8 steps at a time through LDS.
// ---------------------------------------------------------------------------
__global__ __launch_bounds__(DI)
void scan_gate(const float* __restrict__ xz, const float* __restrict__ xc,
               const float* __restrict__ xdbl, const float* __restrict__ dt,
               const float* __restrict__ A_log, const float* __restrict__ Dsk,
               float* __restrict__ ys) {
    const int seq = blockIdx.x;
    const int d   = threadIdx.x;
    __shared__ float sBC[8][2 * DS];

    float Ad[DS];
#pragma unroll
    for (int s = 0; s < DS; ++s) Ad[s] = -expf(A_log[d * DS + s]);
    const float Dd = Dsk[d];
    float h[DS] = {};
    const size_t base = (size_t)seq * NTOK;
    const int c = threadIdx.x >> 5, e = threadIdx.x & 31;

    for (int n0 = 0; n0 < NTOK; n0 += 8) {
        int nn = NTOK - n0; if (nn > 8) nn = 8;
        if (c < nn) sBC[c][e] = xdbl[(base + n0 + c) * LDXD + DTR + e];  // B then C
        __syncthreads();
        for (int cc = 0; cc < nn; ++cc) {
            size_t t = base + n0 + cc;
            float dtv = dt[t * DI + d];
            float xv  = xc[t * DI + d];
            float dx  = dtv * xv;
            float y = 0.f;
#pragma unroll
            for (int s = 0; s < DS; ++s) {
                h[s] = expf(dtv * Ad[s]) * h[s] + dx * sBC[cc][s];
                y += h[s] * sBC[cc][DS + s];
            }
            y += xv * Dd;
            float zg = xz[t * (2 * DI) + DI + d];
            y *= zg / (1.f + expf(-zg));                 // * silu(z-gate)
            ys[t * DI + d] = y;
        }
        __syncthreads();
    }
}

// ---------------------------------------------------------------------------
// Final (N,D) -> (D,N) transpose per sequence via LDS 32x32 tiles
// ---------------------------------------------------------------------------
__global__ void write_out_t(const float* __restrict__ x, float* __restrict__ out,
                            int s0) {
    __shared__ float tile[32][33];
    const int sl = blockIdx.z;
    const int n0 = blockIdx.x * 32, d0 = blockIdx.y * 32;
#pragma unroll
    for (int i = threadIdx.y; i < 32; i += 8) {
        int n = n0 + i;
        if (n < NTOK)
            tile[i][threadIdx.x] = x[((size_t)sl * NTOK + n) * DM + d0 + threadIdx.x];
    }
    __syncthreads();
#pragma unroll
    for (int i = threadIdx.y; i < 32; i += 8) {
        int dmi = d0 + i, n = n0 + threadIdx.x;
        if (n < NTOK)
            out[((size_t)(s0 + sl) * DM + dmi) * NTOK + n] = tile[threadIdx.x][i];
    }
}

// ---------------------------------------------------------------------------
// Host launcher
// ---------------------------------------------------------------------------
static inline int gemm_blocks_nt(int M, int Nv, int NT) {
    int waves = (M / 16) * (Nv / (16 * NT));
    return (waves + 7) / 8;
}
static inline int gemm_blocks(int M, int Nv) {
    int waves = ((M + 15) / 16) * ((Nv + 63) / 64);
    return (waves + 7) / 8;
}

extern "C" void kernel_launch(void* const* d_in, const int* in_sizes, int n_in,
                              void* d_out, int out_size, void* d_ws, size_t ws_size,
                              hipStream_t stream) {
    const float* z        = (const float*)d_in[0];
    const float* proj_w   = (const float*)d_in[1];
    const float* proj_b   = (const float*)d_in[2];
    const float* in_proj  = (const float*)d_in[3];   // L x 128 x 512
    const float* conv_w   = (const float*)d_in[4];
    const float* conv_b   = (const float*)d_in[5];
    const float* x_proj   = (const float*)d_in[6];   // L x 256 x 40
    const float* dtp_w    = (const float*)d_in[7];
    const float* dtp_b    = (const float*)d_in[8];
    const float* A_log    = (const float*)d_in[9];
    const float* D_skip   = (const float*)d_in[10];
    const float* out_proj = (const float*)d_in[11];  // L x 256 x 128
    const float* lin1_w   = (const float*)d_in[12];  // L x 128 x 256
    const float* lin1_b   = (const float*)d_in[13];
    const float* lin2_w   = (const float*)d_in[14];  // L x 256 x 128
    const float* lin2_b   = (const float*)d_in[15];
    float* out = (float*)d_out;

    // f16 weight arena at front of ws, activations after
    _Float16* wh  = (_Float16*)d_ws;
    float*    act = (float*)d_ws + WH_TOTAL / 2;
    const size_t perTok = DM + 2 * DI + DI + LDXD + DI + DI;   // 1456 floats
    size_t wsFloats  = ws_size / sizeof(float);
    size_t actFloats = (wsFloats > (size_t)(WH_TOTAL / 2)) ? wsFloats - WH_TOTAL / 2 : 0;
    int ncMax = (int)(actFloats / ((size_t)NTOK * perTok));
    if (ncMax < 1) ncMax = 1;
    bool full = false;
    if (ncMax >= 16) { ncMax &= ~15; full = true; }  // Mc % 16 == 0 always (224 = 14*16)
    if (ncMax > NSEQ) ncMax = NSEQ;

    // one-time weight conversion to f16 col-major (x_proj padded 40 -> 48)
    for (int l = 0; l < 3; ++l) {
        _Float16* whl = wh + (size_t)l * WH_LAYER;
        convert_w<<<(128 * 512 + 255) / 256, 256, 0, stream>>>(in_proj  + (size_t)l * 65536, whl,          128, 512, 512);
        convert_w<<<(256 * 128 + 255) / 256, 256, 0, stream>>>(out_proj + (size_t)l * 32768, whl + 65536,  256, 128, 128);
        convert_w<<<(128 * 256 + 255) / 256, 256, 0, stream>>>(lin1_w   + (size_t)l * 32768, whl + 98304,  128, 256, 256);
        convert_w<<<(256 * 128 + 255) / 256, 256, 0, stream>>>(lin2_w   + (size_t)l * 32768, whl + 131072, 256, 128, 128);
        convert_w<<<(256 * 48  + 255) / 256, 256, 0, stream>>>(x_proj   + (size_t)l * 10240, whl + 163840, 256, 40, 48);
    }

    float* bx   = act;
    float* bxz  = bx   + (size_t)ncMax * NTOK * DM;
    float* bxc  = bxz  + (size_t)ncMax * NTOK * 2 * DI;
    float* bxd  = bxc  + (size_t)ncMax * NTOK * DI;
    float* bdt  = bxd  + (size_t)ncMax * NTOK * LDXD;
    float* bys  = bdt  + (size_t)ncMax * NTOK * DI;

    for (int s0 = 0; s0 < NSEQ; s0 += ncMax) {
        int nc = (NSEQ - s0 < ncMax) ? (NSEQ - s0) : ncMax;
        int Mc = nc * NTOK;
        int eltB = (Mc * DI + 255) / 256;

        proj_wmma<<<gemm_blocks(Mc, DM), 256, 0, stream>>>(z, proj_w, proj_b, bx, s0, Mc);

        for (int l = 0; l < 3; ++l) {
            const _Float16* whl = wh + (size_t)l * WH_LAYER;
            // xz = x @ in_proj
            if (full)
                gemm_wmma_fast<0, 128, 4><<<gemm_blocks_nt(Mc, 2 * DI, 4), 256, 0, stream>>>(
                    bx, whl, nullptr, bxz, Mc, 2 * DI, DM, 2 * DI);
            else
                gemm_wmma<0><<<gemm_blocks(Mc, 2 * DI), 256, 0, stream>>>(
                    bx, in_proj + (size_t)l * 65536, nullptr, bxz,
                    Mc, 2 * DI, DM, DM, 2 * DI, 2 * DI);
            // xc = silu(causal_dw_conv(xz[:, :256]))
            conv_silu<<<eltB, 256, 0, stream>>>(bxz, conv_w + (size_t)l * DI * 4,
                                                conv_b + (size_t)l * DI, bxc, Mc);
            // x_dbl = xc @ x_proj  (padded f16 path: Nv=48, NT=3)
            if (full)
                gemm_wmma_fast<0, 256, 3><<<gemm_blocks_nt(Mc, 48, 3), 256, 0, stream>>>(
                    bxc, whl + 163840, nullptr, bxd, Mc, 48, DI, LDXD);
            else
                gemm_wmma<0><<<gemm_blocks(Mc, 40), 256, 0, stream>>>(
                    bxc, x_proj + (size_t)l * 10240, nullptr, bxd,
                    Mc, 40, DI, DI, 40, LDXD);
            // dt = softplus(x_dbl[:, :8] @ dt_proj + b)
            dt_softplus<<<eltB, 256, 0, stream>>>(bxd, dtp_w + (size_t)l * DTR * DI,
                                                  dtp_b + (size_t)l * DI, bdt, Mc);
            // selective scan + D-skip + silu(z) gating
            scan_gate<<<nc, DI, 0, stream>>>(bxz, bxc, bxd, bdt,
                                             A_log + (size_t)l * DI * DS,
                                             D_skip + (size_t)l * DI, bys);
            // y2 = ys @ out_proj   (reuse bxz)
            if (full)
                gemm_wmma_fast<0, 256, 4><<<gemm_blocks_nt(Mc, DM, 4), 256, 0, stream>>>(
                    bys, whl + 65536, nullptr, bxz, Mc, DM, DI, DM);
            else
                gemm_wmma<0><<<gemm_blocks(Mc, DM), 256, 0, stream>>>(
                    bys, out_proj + (size_t)l * 32768, nullptr, bxz,
                    Mc, DM, DI, DI, DM, DM);
            // h = gelu(y2 @ lin1 + b)  (reuse bxc)
            if (full)
                gemm_wmma_fast<2, 128, 4><<<gemm_blocks_nt(Mc, DFF, 4), 256, 0, stream>>>(
                    bxz, whl + 98304, lin1_b + (size_t)l * DFF, bxc, Mc, DFF, DM, DFF);
            else
                gemm_wmma<2><<<gemm_blocks(Mc, DFF), 256, 0, stream>>>(
                    bxz, lin1_w + (size_t)l * 32768, lin1_b + (size_t)l * DFF, bxc,
                    Mc, DFF, DM, DM, DFF, DFF);
            // x = h @ lin2 + b
            if (full)
                gemm_wmma_fast<1, 256, 4><<<gemm_blocks_nt(Mc, DM, 4), 256, 0, stream>>>(
                    bxc, whl + 131072, lin2_b + (size_t)l * DM, bx, Mc, DM, DFF, DM);
            else
                gemm_wmma<1><<<gemm_blocks(Mc, DM), 256, 0, stream>>>(
                    bxc, lin2_w + (size_t)l * 32768, lin2_b + (size_t)l * DM, bx,
                    Mc, DM, DFF, DFF, DM, DM);
        }
        write_out_t<<<dim3((NTOK + 31) / 32, DM / 32, nc), dim3(32, 8), 0, stream>>>(bx, out, s0);
    }
}
